// Encoder_z0_ODE_RNN_77996606095865
// MI455X (gfx1250) — compile-verified, matrix operand order checked by reading, not run on hardware
//
#include <hip/hip_runtime.h>
#include <math.h>

// ---------------------------------------------------------------------------
// GRU-ODE encoder for MI455X (gfx1250), wave32 + WMMA bf16.
// Each workgroup owns a 16-row batch tile (M = WMMA tile) and runs all 200
// scan steps with y/s state in LDS. Weights pre-packed to bf16 WMMA-B
// fragments in d_ws (3.3 MB, L2-resident). Dual-tile WMMA per wave for ILP,
// with unroll caps to stay under 256 VGPRs (no spills, no vgpr-msb churn).
// ---------------------------------------------------------------------------

typedef __attribute__((ext_vector_type(16))) __bf16 v16bf;
typedef __attribute__((ext_vector_type(8)))  __bf16 v8bf;
typedef __attribute__((ext_vector_type(8)))  float  v8f;

#define B_TOT  2048
#define NT_    200
#define IN_DIM 128
#define LAT_   256
#define BT     16      // batch rows per workgroup (= WMMA M)
#define NWAVES 8       // 256 threads

// packed bf16 weight offsets inside d_ws (in elements)
#define O_WO1 0u
#define O_WO2 65536u
#define O_WU1 131072u
#define O_WU2 458752u
#define O_WR1 589824u
#define O_WR2 917504u
#define O_WN1 1048576u
#define O_WN2 1376256u

// LDS carve (bytes)
#define SM_BYTES 127488

__device__ __forceinline__ unsigned short f2bf(float f) {
  unsigned int u = __float_as_uint(f);
  u += 0x7FFFu + ((u >> 16) & 1u);           // round-to-nearest-even
  return (unsigned short)(u >> 16);
}
__device__ __forceinline__ unsigned int pk2(float a, float b) {
  return (unsigned int)f2bf(a) | ((unsigned int)f2bf(b) << 16);
}
// branch-free saturating tanh / sigmoid on v_exp_f32 + v_rcp_f32
__device__ __forceinline__ float fast_tanh(float x) {
  float e = __expf(2.0f * x);                // inf for large x -> rcp -> 0 -> 1
  return 1.0f - 2.0f * __builtin_amdgcn_rcpf(e + 1.0f);
}
__device__ __forceinline__ float sigm(float x) {
  return __builtin_amdgcn_rcpf(1.0f + __expf(-x));
}

// ---------------------------------------------------------------------------
// Pre-pack fp32 row-major W[K][N] into bf16 WMMA-B fragments: fragment
// f = (ntile, ktile) is a 32x16 (KxN) block stored as 32 lanes x 16 bf16
// contiguous, matching the 16-bit operand VGPR striping.
// ---------------------------------------------------------------------------
__global__ void prepack_wmma_b_kernel(const float* __restrict__ W,
                                      unsigned short* __restrict__ dst,
                                      int K, int N) {
  const int f    = blockIdx.x;
  const int nkt  = K >> 5;
  const int ntl  = f / nkt;
  const int ktl  = f % nkt;
  const int lane = threadIdx.x;
  const int n    = ntl * 16 + (lane & 15);
#pragma unroll
  for (int i = 0; i < 16; ++i) {
    int ko = (lane < 16) ? ((i < 8) ? i : i + 8) : ((i < 8) ? i + 8 : i + 16);
    int k  = ktl * 32 + ko;
    dst[(size_t)f * 512 + lane * 16 + i] = f2bf(W[(size_t)k * N + n]);
  }
}

// Two 16x16 output tiles (nt0, nt1) of act[16 x K] (bf16 LDS, row stride
// strideE) x packed weights, sharing the A fragment: 2 independent WMMA
// accumulation chains. kt-loop unroll capped at 2 to bound VGPR pressure.
__device__ __forceinline__ void wmma_row2(const unsigned short* __restrict__ act,
                                          int strideE, int K,
                                          const unsigned short* __restrict__ wfrag,
                                          int nt0, int nt1, int lane,
                                          v8f& acc0, v8f& acc1) {
  const int nkt  = K >> 5;
  const int row  = lane & 15;
  const int asel = (lane < 16) ? 0 : 8;
  const v16bf* wb = (const v16bf*)wfrag;
  v8f c0 = {0.f, 0.f, 0.f, 0.f, 0.f, 0.f, 0.f, 0.f};
  v8f c1 = {0.f, 0.f, 0.f, 0.f, 0.f, 0.f, 0.f, 0.f};
#pragma unroll 2
  for (int kt = 0; kt < nkt; ++kt) {
    const unsigned short* ap = act + row * strideE + kt * 32 + asel;
    v8bf lo = *(const v8bf*)(ap);        // ds_load_b128
    v8bf hi = *(const v8bf*)(ap + 16);   // ds_load_b128
    v16bf a;
#pragma unroll
    for (int i = 0; i < 8; ++i) { a[i] = lo[i]; a[i + 8] = hi[i]; }
    v16bf b0 = wb[(size_t)(nt0 * nkt + kt) * 32 + lane];
    v16bf b1 = wb[(size_t)(nt1 * nkt + kt) * 32 + lane];
    c0 = __builtin_amdgcn_wmma_f32_16x16x32_bf16(false, a, false, b0,
                                                 (short)0, c0, false, false);
    c1 = __builtin_amdgcn_wmma_f32_16x16x32_bf16(false, a, false, b1,
                                                 (short)0, c1, false, false);
  }
  acc0 = c0; acc1 = c1;
}

// ---------------------------------------------------------------------------
// Persistent per-batch-tile scan kernel. grid = 128 blocks x 256 threads.
// ---------------------------------------------------------------------------
__global__ void __launch_bounds__(256)
enc_ode_rnn_kernel(const float* __restrict__ data, const float* __restrict__ ts,
                   const unsigned short* __restrict__ wp,
                   const float* __restrict__ bu1, const float* __restrict__ bu2,
                   const float* __restrict__ br1, const float* __restrict__ br2,
                   const float* __restrict__ bn1, const float* __restrict__ bn2,
                   const float* __restrict__ bo1, const float* __restrict__ bo2,
                   const float* __restrict__ Wt1, const float* __restrict__ bt1,
                   const float* __restrict__ Wt2, const float* __restrict__ bt2,
                   float* __restrict__ out) {
  extern __shared__ char smem[];
  float*          sy  = (float*)smem;                       // 16x256 f32 state y
  float*          ss  = sy  + BT * LAT_;                    // 16x256 f32 state std
  float*          syo = ss  + BT * LAT_;                    // 16x256 f32 y_ode
  float*          su  = syo + BT * LAT_;                    // 16x256 f32 gate u
  float*          sr  = su  + BT * LAT_;                    // 16x256 f32 gate r
  float*          smk = sr  + BT * LAT_;                    // 16 f32 obs mask
  unsigned short* yc  = (unsigned short*)(smk + 16);        // 16x640 bf16 concat
  unsigned short* sh  = yc + BT * 640;                      // 16x512 bf16 hidden
  unsigned short* yb  = sh + BT * 512;                      // 16x256 bf16 y
  float*          shf = (float*)(yb + BT * LAT_);           // 112 f32 final hidden

  const int tid  = threadIdx.x;
  const int lane = tid & 31;
  const int wv   = tid >> 5;
  const int b0   = blockIdx.x * BT;

  // epilogue helpers -------------------------------------------------------
  auto ep_tanh_sh = [&](const v8f& a, int nt, const float* __restrict__ bias) {
    int c = nt * 16 + (lane & 15); float bv = bias[c];
#pragma unroll
    for (int j = 0; j < 8; ++j) {
      int row = (lane < 16) ? j : j + 8;
      sh[row * 512 + c] = f2bf(fast_tanh(a[j] + bv));
    }
  };
  auto ep_sig = [&](const v8f& a, int nt, const float* __restrict__ bias,
                    float* __restrict__ dstb) {
    int c = nt * 16 + (lane & 15); float bv = bias[c];
#pragma unroll
    for (int j = 0; j < 8; ++j) {
      int row = (lane < 16) ? j : j + 8;
      dstb[row * 256 + c] = sigm(a[j] + bv);
    }
  };

  for (int i = tid; i < BT * LAT_; i += 256) { sy[i] = 0.f; ss[i] = 0.f; }
  __syncthreads();

  for (int ti = 0; ti < NT_; ++ti) {
    const int   tt = NT_ - 1 - ti;
    const float dt = (ti == 0) ? -0.01f : (ts[tt] - ts[tt + 1]);

    // ---- phase A: bf16 y, bf16 s -> yc[:,256:512], x -> yc[:,512:640], mask
    for (int i4 = tid; i4 < BT * LAT_ / 4; i4 += 256) {
      int i = i4 * 4;
      float4 v = *(const float4*)(sy + i);
      *(uint2*)(yb + i) = make_uint2(pk2(v.x, v.y), pk2(v.z, v.w));
    }
    for (int i4 = tid; i4 < BT * LAT_ / 4; i4 += 256) {
      int i = i4 * 4, r = i >> 8, c = i & 255;
      float4 v = *(const float4*)(ss + i);
      *(uint2*)(yc + r * 640 + 256 + c) = make_uint2(pk2(v.x, v.y), pk2(v.z, v.w));
    }
    for (int i4 = tid; i4 < BT * IN_DIM / 4; i4 += 256) {
      int i = i4 * 4, r = i >> 7, c = i & 127;
      float4 v = *(const float4*)(&data[((size_t)(b0 + r) * NT_ + tt) * IN_DIM + c]);
      *(uint2*)(yc + r * 640 + 512 + c) = make_uint2(pk2(v.x, v.y), pk2(v.z, v.w));
    }
    if (tid < BT) {
      const float4* xp =
          (const float4*)&data[((size_t)(b0 + tid) * NT_ + tt) * IN_DIM + 64];
      float sum = 0.f;
#pragma unroll
      for (int k = 0; k < 16; ++k) {
        float4 v = xp[k];
        sum += v.x + v.y + v.z + v.w;
      }
      smk[tid] = (sum > 0.f) ? 1.f : 0.f;
    }
    __syncthreads();

    // ---- G1: h1 = tanh(y @ Wo1 + bo1)          [K=256, N=256] -> sh
    {
      v8f a0, a1;
      wmma_row2(yb, 256, 256, wp + O_WO1, wv, wv + 8, lane, a0, a1);
      ep_tanh_sh(a0, wv, bo1); ep_tanh_sh(a1, wv + 8, bo1);
    }
    __syncthreads();

    // ---- G2: y_ode = y + (h1 @ Wo2 + bo2)*dt   [K=256, N=256]
    {
      v8f a0, a1;
      wmma_row2(sh, 512, 256, wp + O_WO2, wv, wv + 8, lane, a0, a1);
      auto ep = [&](const v8f& a, int nt) {
        int c = nt * 16 + (lane & 15); float bv = bo2[c];
#pragma unroll
        for (int j = 0; j < 8; ++j) {
          int row = (lane < 16) ? j : j + 8;
          float yo = sy[row * 256 + c] + (a[j] + bv) * dt;
          syo[row * 256 + c] = yo;
          yc[row * 640 + c]  = f2bf(yo);
        }
      };
      ep(a0, wv); ep(a1, wv + 8);
    }
    __syncthreads();

    // ---- G3: hu = tanh(yc @ Wu1 + bu1)         [K=640, N=512] -> sh
#pragma unroll 1
    for (int p = 0; p < 2; ++p) {
      int base = wv + p * 16;
      v8f a0, a1;
      wmma_row2(yc, 640, 640, wp + O_WU1, base, base + 8, lane, a0, a1);
      ep_tanh_sh(a0, base, bu1); ep_tanh_sh(a1, base + 8, bu1);
    }
    __syncthreads();

    // ---- G4: u = sigmoid(hu @ Wu2 + bu2)       [K=512, N=256] -> su
    {
      v8f a0, a1;
      wmma_row2(sh, 512, 512, wp + O_WU2, wv, wv + 8, lane, a0, a1);
      ep_sig(a0, wv, bu2, su); ep_sig(a1, wv + 8, bu2, su);
    }
    __syncthreads();

    // ---- G5: hr = tanh(yc @ Wr1 + br1)         [K=640, N=512] -> sh
#pragma unroll 1
    for (int p = 0; p < 2; ++p) {
      int base = wv + p * 16;
      v8f a0, a1;
      wmma_row2(yc, 640, 640, wp + O_WR1, base, base + 8, lane, a0, a1);
      ep_tanh_sh(a0, base, br1); ep_tanh_sh(a1, base + 8, br1);
    }
    __syncthreads();

    // ---- G6: r = sigmoid(hr @ Wr2 + br2)       [K=512, N=256] -> sr
    {
      v8f a0, a1;
      wmma_row2(sh, 512, 512, wp + O_WR2, wv, wv + 8, lane, a0, a1);
      ep_sig(a0, wv, br2, sr); ep_sig(a1, wv + 8, br2, sr);
    }
    __syncthreads();

    // ---- c2 = [y_ode*r, s*r, x] (overwrite yc in place; x untouched)
    for (int i4 = tid; i4 < BT * LAT_ / 4; i4 += 256) {
      int i = i4 * 4, r = i >> 8, c = i & 255;
      float4 vy = *(const float4*)(syo + i);
      float4 vr = *(const float4*)(sr + i);
      float4 vs = *(const float4*)(ss + i);
      *(uint2*)(yc + r * 640 + c) =
          make_uint2(pk2(vy.x * vr.x, vy.y * vr.y), pk2(vy.z * vr.z, vy.w * vr.w));
      *(uint2*)(yc + r * 640 + 256 + c) =
          make_uint2(pk2(vs.x * vr.x, vs.y * vr.y), pk2(vs.z * vr.z, vs.w * vr.w));
    }
    __syncthreads();

    // ---- G7: hn = tanh(c2 @ Wn1 + bn1)         [K=640, N=512] -> sh
#pragma unroll 1
    for (int p = 0; p < 2; ++p) {
      int base = wv + p * 16;
      v8f a0, a1;
      wmma_row2(yc, 640, 640, wp + O_WN1, base, base + 8, lane, a0, a1);
      ep_tanh_sh(a0, base, bn1); ep_tanh_sh(a1, base + 8, bn1);
    }
    __syncthreads();

    // ---- G8: ns = hn @ Wn2 + bn2 + GRU gating/mask, write y,s in place
    {
      auto ep_gate = [&](const v8f& a, int nt) {
        int c = nt * 16 + (lane & 15); float bv = bn2[c];
#pragma unroll
        for (int j = 0; j < 8; ++j) {
          int row = (lane < 16) ? j : j + 8;
          float nsv = a[j] + bv;
          float m   = smk[row];
          if (c < 256) {                      // uniform per wave (nt uniform)
            float uu = su[row * 256 + c];
            float yo = syo[row * 256 + c];
            float ny = (1.f - uu) * nsv + uu * yo;
            sy[row * 256 + c] = m * ny + (1.f - m) * yo;
          } else {
            int cc = c - 256;
            float uu = su[row * 256 + cc];
            float so = ss[row * 256 + cc];
            float pre = (1.f - uu) * fabsf(nsv) + uu * so;
            ss[row * 256 + cc] = fabsf(m * pre + (1.f - m) * so);
          }
        }
      };
#pragma unroll 1
      for (int p = 0; p < 2; ++p) {
        int base = wv + p * 16;
        v8f a0, a1;
        wmma_row2(sh, 512, 512, wp + O_WN2, base, base + 8, lane, a0, a1);
        ep_gate(a0, base); ep_gate(a1, base + 8);
      }
    }
    __syncthreads();
  }

  // ---- final MLP: z = tanh([y,s] @ Wt1 + bt1) @ Wt2 + bt2  (K=100, VALU)
  for (int r = 0; r < BT; ++r) {
    for (int j = tid; j < 100; j += 256) {
      float sum = bt1[j];
      for (int k = 0; k < 256; ++k) sum += sy[r * 256 + k] * Wt1[(size_t)k * 100 + j];
      for (int k = 0; k < 256; ++k) sum += ss[r * 256 + k] * Wt1[(size_t)(256 + k) * 100 + j];
      shf[j] = fast_tanh(sum);
    }
    __syncthreads();
    for (int c = tid; c < 512; c += 256) {
      float sum = bt2[c];
      for (int j = 0; j < 100; ++j) sum += shf[j] * Wt2[(size_t)j * 512 + c];
      int b = b0 + r;
      if (c < 256) out[(size_t)b * 256 + c] = sum;                               // mean_z0
      else out[(size_t)B_TOT * 256 + (size_t)b * 256 + (c - 256)] =
               fmaxf(fabsf(sum), 1e-20f);                                        // std_z0
    }
    __syncthreads();
  }
}

// ---------------------------------------------------------------------------
extern "C" void kernel_launch(void* const* d_in, const int* in_sizes, int n_in,
                              void* d_out, int out_size, void* d_ws, size_t ws_size,
                              hipStream_t stream) {
  const float* data = (const float*)d_in[0];
  const float* ts   = (const float*)d_in[1];
  const float* Wu1  = (const float*)d_in[2];  const float* bu1 = (const float*)d_in[3];
  const float* Wu2  = (const float*)d_in[4];  const float* bu2 = (const float*)d_in[5];
  const float* Wr1  = (const float*)d_in[6];  const float* br1 = (const float*)d_in[7];
  const float* Wr2  = (const float*)d_in[8];  const float* br2 = (const float*)d_in[9];
  const float* Wn1  = (const float*)d_in[10]; const float* bn1 = (const float*)d_in[11];
  const float* Wn2  = (const float*)d_in[12]; const float* bn2 = (const float*)d_in[13];
  const float* Wo1  = (const float*)d_in[14]; const float* bo1 = (const float*)d_in[15];
  const float* Wo2  = (const float*)d_in[16]; const float* bo2 = (const float*)d_in[17];
  const float* Wt1  = (const float*)d_in[18]; const float* bt1 = (const float*)d_in[19];
  const float* Wt2  = (const float*)d_in[20]; const float* bt2 = (const float*)d_in[21];
  float* out = (float*)d_out;
  unsigned short* wp = (unsigned short*)d_ws;

  // pre-pack the 8 recurrent weight matrices to bf16 WMMA-B fragments
  prepack_wmma_b_kernel<<<dim3(8  * 16), 32, 0, stream>>>(Wo1, wp + O_WO1, 256, 256);
  prepack_wmma_b_kernel<<<dim3(8  * 16), 32, 0, stream>>>(Wo2, wp + O_WO2, 256, 256);
  prepack_wmma_b_kernel<<<dim3(20 * 32), 32, 0, stream>>>(Wu1, wp + O_WU1, 640, 512);
  prepack_wmma_b_kernel<<<dim3(16 * 16), 32, 0, stream>>>(Wu2, wp + O_WU2, 512, 256);
  prepack_wmma_b_kernel<<<dim3(20 * 32), 32, 0, stream>>>(Wr1, wp + O_WR1, 640, 512);
  prepack_wmma_b_kernel<<<dim3(16 * 16), 32, 0, stream>>>(Wr2, wp + O_WR2, 512, 256);
  prepack_wmma_b_kernel<<<dim3(20 * 32), 32, 0, stream>>>(Wn1, wp + O_WN1, 640, 512);
  prepack_wmma_b_kernel<<<dim3(16 * 32), 32, 0, stream>>>(Wn2, wp + O_WN2, 512, 512);

  // persistent scan: one WG per 16 batch rows
  enc_ode_rnn_kernel<<<dim3(B_TOT / BT), 256, SM_BYTES, stream>>>(
      data, ts, wp, bu1, bu2, br1, br2, bn1, bn2, bo1, bo2,
      Wt1, bt1, Wt2, bt2, out);
}
